// LoRaMoELinear_26482768347855
// MI455X (gfx1250) — matrix operand order repaired and checked by reading, not compiled, not run on hardware
//
#include <hip/hip_runtime.h>
#include <stdint.h>

// ---------------- problem constants ----------------
#define N_TOK 8192
#define DIN   4096
#define DOUT  4096
#define NE    8
#define RANK  16
#define ER    128          // NE * RANK
#define H4E   32           // 4*E gate hidden
#define CAP   3072         // int(3.0 * N / E)
#define ALPHA (1.0f/16.0f)
#define LN_EPS 1e-5f

// ---------------- GEMM tiling ----------------
#define BM 128
#define BN 128
#define BK 32
#define XSTR 40            // padded LDS row stride in ushort (80B: 16B-aligned, conflict-free)
#define NKT_DENSE (DIN/BK) // 128
#define NKT_ALL   (DIN/BK + ER/BK) // 132

typedef __attribute__((ext_vector_type(8)))  __bf16 v8bf;
typedef __attribute__((ext_vector_type(16))) __bf16 v16bf;
typedef __attribute__((ext_vector_type(8)))  float  v8f;
typedef __attribute__((ext_vector_type(4)))  int    v4i;

#define AS1 __attribute__((address_space(1)))
#define AS3 __attribute__((address_space(3)))

__device__ __forceinline__ unsigned short f2bf(float f) {
  unsigned u = __float_as_uint(f);
  unsigned r = 0x7FFFu + ((u >> 16) & 1u);   // round-to-nearest-even
  return (unsigned short)((u + r) >> 16);
}

// ---- CDNA5 async global->LDS copy (ASYNCcnt-tracked), 16B per lane ----
__device__ __forceinline__ void async_cp16(const unsigned short* g, unsigned short* l) {
#if __has_builtin(__builtin_amdgcn_global_load_async_to_lds_b128)
  __builtin_amdgcn_global_load_async_to_lds_b128((AS1 v4i*)g, (AS3 v4i*)l, 0, 0);
#else
  unsigned loff = (unsigned)(size_t)(AS3 void*)l;
  asm volatile("global_load_async_to_lds_b128 %0, %1, off" :: "v"(loff), "v"(g) : "memory");
#endif
}

#if __has_builtin(__builtin_amdgcn_s_wait_asynccnt)
#define WAIT_ASYNC(n) __builtin_amdgcn_s_wait_asynccnt(n)
#else
#define WAIT_ASYNC(n) asm volatile("s_wait_asynccnt %0" :: "i"(n) : "memory")
#endif

// Build a v16bf WMMA fragment from two 16B chunks in LDS (gap in ushorts).
__device__ __forceinline__ v16bf load_frag(const unsigned short* p, int gap) {
  v8bf lo = *(const v8bf*)p;
  v8bf hi = *(const v8bf*)(p + gap);
  return __builtin_shufflevector(lo, hi, 0,1,2,3,4,5,6,7,8,9,10,11,12,13,14,15);
}

// Shared WMMA micro-tile: 2x4 tiles of 16x16, K=32 step.
__device__ __forceinline__ void tile_mma(const unsigned short* xs, const unsigned short* wsm,
                                         int waveM, int waveN, int l15, int sel,
                                         v8f acc[2][4]) {
  v16bf afr[2], bfr[4];
  #pragma unroll
  for (int tm = 0; tm < 2; ++tm) {
    int m = waveM*32 + tm*16 + l15;
    // A 16x32 bf16 layout: lanes<16 K0..7 & K16..23; lanes>=16 K8..15 & K24..31
    afr[tm] = load_frag(&xs[m*XSTR + sel*8], 16);
  }
  #pragma unroll
  for (int tn = 0; tn < 4; ++tn) {
    int nn = waveN*64 + tn*16 + l15;
    // B 32x16 bf16 layout: lanes<16 K0..7 & K16..23; lanes>=16 K8..15 & K24..31
    bfr[tn] = load_frag(&wsm[nn*XSTR + sel*8], 16);
  }
  #pragma unroll
  for (int tm = 0; tm < 2; ++tm)
    #pragma unroll
    for (int tn = 0; tn < 4; ++tn)
      acc[tm][tn] = __builtin_amdgcn_wmma_f32_16x16x32_bf16(
          false, afr[tm], false, bfr[tn], (short)0, acc[tm][tn], false, false);
}

// ---------------- gate MLP: one wave per token ----------------
__global__ void gate_kernel(const float* __restrict__ x,
                            const float* __restrict__ gw1,
                            const float* __restrict__ gb1,
                            const float* __restrict__ lng,
                            const float* __restrict__ lnb,
                            const float* __restrict__ gw2,
                            const float* __restrict__ gb2,
                            float* __restrict__ gates) {
  int lane = threadIdx.x & 31;
  int wave = threadIdx.x >> 5;
  int n = blockIdx.x * 8 + wave;
  const float4* xr = (const float4*)(x + (size_t)n * DIN);
  const float4* gr = (const float4*)(gw1 + (size_t)lane * DIN);
  float acc = 0.f;
  for (int i = 0; i < DIN/4; ++i) {
    float4 a = xr[i], b = gr[i];
    acc += a.x*b.x + a.y*b.y + a.z*b.z + a.w*b.w;
  }
  float h = acc + gb1[lane];
  float s = h;
  #pragma unroll
  for (int off = 16; off; off >>= 1) s += __shfl_xor(s, off);
  float mu = s * (1.0f/32.0f);
  float d  = h - mu;
  float v  = d * d;
  #pragma unroll
  for (int off = 16; off; off >>= 1) v += __shfl_xor(v, off);
  v *= (1.0f/32.0f);
  float hn = d * rsqrtf(v + LN_EPS) * lng[lane] + lnb[lane];
  hn = fmaxf(hn, 0.f);
  float gv = 0.f;
  for (int e = 0; e < NE; ++e) {
    float t = hn * gw2[e*H4E + lane];
    #pragma unroll
    for (int off = 16; off; off >>= 1) t += __shfl_xor(t, off);
    if (lane == e) gv = t + gb2[e];
  }
  if (lane < NE) gates[(size_t)n*NE + lane] = gv;
}

// ---------------- routing: top-2 + softmax + expert counts ----------------
__global__ void zero_counts_kernel(int* __restrict__ counts) {
  if (threadIdx.x < 2*NE) counts[threadIdx.x] = 0;
}

__global__ void route_kernel(const float* __restrict__ gates,
                             int* __restrict__ topi, float* __restrict__ topg,
                             int* __restrict__ counts) {
  int n = blockIdx.x * blockDim.x + threadIdx.x;
  if (n >= N_TOK) return;
  float g[NE];
  #pragma unroll
  for (int e = 0; e < NE; ++e) g[e] = gates[(size_t)n*NE + e];
  int i1 = 0; float v1 = g[0];
  #pragma unroll
  for (int e = 1; e < NE; ++e) if (g[e] > v1) { v1 = g[e]; i1 = e; }
  int i2 = -1; float v2 = -3.4e38f;
  #pragma unroll
  for (int e = 0; e < NE; ++e) if (e != i1 && g[e] > v2) { v2 = g[e]; i2 = e; }
  float m  = fmaxf(v1, v2);
  float e1 = __expf(v1 - m), e2 = __expf(v2 - m);
  float inv = 1.0f / (e1 + e2);
  topi[n*2]   = i1; topi[n*2+1] = i2;
  topg[n*2]   = e1*inv; topg[n*2+1] = e2*inv;
  atomicAdd(&counts[0*NE + i1], 1);
  atomicAdd(&counts[1*NE + i2], 1);
}

// ---------------- per-token combine weights with capacity rule ----------------
__global__ void wcomb_kernel(const int* __restrict__ topi,
                             const float* __restrict__ topg,
                             const int* __restrict__ counts,
                             float* __restrict__ w) {
  int n = blockIdx.x * blockDim.x + threadIdx.x;
  if (n >= N_TOK) return;
  float wv[NE];
  #pragma unroll
  for (int e = 0; e < NE; ++e) wv[e] = 0.f;
  #pragma unroll
  for (int k = 0; k < 2; ++k) {
    int e = topi[n*2+k];
    if (counts[k*NE + e] <= CAP) wv[e] += topg[n*2+k];
  }
  #pragma unroll
  for (int e = 0; e < NE; ++e) w[(size_t)n*NE + e] = wv[e];
}

// ---------------- LoRA down-proj: Tg[n, e*R+r] = (x . A[e,:,r]) * w[n,e] * alpha ----------------
__global__ void lora_t_kernel(const float* __restrict__ x,
                              const float* __restrict__ lora_A,
                              const float* __restrict__ w,
                              unsigned short* __restrict__ Tg) {
  __shared__ float xs[DIN];
  int n = blockIdx.x;
  int t = threadIdx.x;          // 0..127
  const float4* xr = (const float4*)(x + (size_t)n * DIN);
  float4* xs4 = (float4*)xs;
  for (int i = t; i < DIN/4; i += 128) xs4[i] = xr[i];
  __syncthreads();
  int e = t >> 4, r = t & 15;
  float wv = w[(size_t)n*NE + e];
  float acc = 0.f;
  if (wv != 0.f) {
    const float* A = lora_A + (size_t)e*DIN*RANK + r;
    for (int i = 0; i < DIN; ++i) acc += xs[i] * A[(size_t)i*RANK];
  }
  Tg[(size_t)n*ER + t] = f2bf(acc * wv * ALPHA);
}

// ---------------- lora_B [ER, OUT] fp32 -> transposed bf16 BcT [OUT, ER] ----------------
__global__ void bconvT_kernel(const float* __restrict__ lora_B,
                              unsigned short* __restrict__ BcT) {
  int idx = blockIdx.x * blockDim.x + threadIdx.x;   // over ER*DOUT, coalesced read
  if (idx >= ER*DOUT) return;
  int k = idx >> 12;            // / DOUT
  int o = idx & (DOUT-1);
  BcT[(size_t)o*ER + k] = f2bf(lora_B[idx]);
}

// ---------------- fp32 -> bf16 bulk converts (8 elems / thread) ----------------
__global__ void cvt_bf16_kernel(const float* __restrict__ src,
                                unsigned short* __restrict__ dst) {
  size_t i = ((size_t)blockIdx.x * blockDim.x + threadIdx.x) * 8;
  float4 a = *(const float4*)(src + i);
  float4 b = *(const float4*)(src + i + 4);
  uint4 o;
  o.x = (unsigned)f2bf(a.x) | ((unsigned)f2bf(a.y) << 16);
  o.y = (unsigned)f2bf(a.z) | ((unsigned)f2bf(a.w) << 16);
  o.z = (unsigned)f2bf(b.x) | ((unsigned)f2bf(b.y) << 16);
  o.w = (unsigned)f2bf(b.z) | ((unsigned)f2bf(b.w) << 16);
  *(uint4*)(dst + i) = o;
}

// ======================================================================
// FAST PATH: bf16 operands in ws; async double-buffered global->LDS copies
// ======================================================================
__global__ __launch_bounds__(256) void main_gemm_async_kernel(
    const unsigned short* __restrict__ xb,  // [N_TOK, DIN]  bf16
    const unsigned short* __restrict__ wb,  // [DOUT,  DIN]  bf16
    const unsigned short* __restrict__ Tg,  // [N_TOK, ER]   bf16 (gated)
    const unsigned short* __restrict__ BcT, // [DOUT,  ER]   bf16
    float* __restrict__ out) {
  __shared__ unsigned short xs [2][BM * XSTR];
  __shared__ unsigned short wsm[2][BN * XSTR];

  int tid  = threadIdx.x;
  int lane = tid & 31, wave = tid >> 5;
  int waveM = wave & 3, waveN = wave >> 2;
  int l15 = lane & 15, sel = lane >> 4;
  int n0 = blockIdx.y * BM, o0 = blockIdx.x * BN;

  v8f acc[2][4] = {};

  // issue one K-step tile (x + W) into buffer `buf`: 4 async b128 per thread
  auto issue = [&](int kt, int buf) {
    const unsigned short* sx; const unsigned short* sw; int stride;
    if (kt < NKT_DENSE) {
      int k0 = kt * BK;
      sx = xb  + (size_t)n0 * DIN + k0;
      sw = wb  + (size_t)o0 * DIN + k0;
      stride = DIN;
    } else {
      int k0 = (kt - NKT_DENSE) * BK;
      sx = Tg  + (size_t)n0 * ER + k0;
      sw = BcT + (size_t)o0 * ER + k0;
      stride = ER;
    }
    #pragma unroll
    for (int j = 0; j < 2; ++j) {
      int idx = j*256 + tid;      // 0..511 lane-chunks of 8 bf16
      int m = idx >> 2, c = idx & 3;
      async_cp16(sx + (size_t)m*stride + c*8, &xs [buf][m*XSTR + c*8]);
      async_cp16(sw + (size_t)m*stride + c*8, &wsm[buf][m*XSTR + c*8]);
    }
  };

  issue(0, 0);
  for (int kt = 0; kt < NKT_ALL; ++kt) {
    int buf = kt & 1;
    if (kt + 1 < NKT_ALL) {
      issue(kt + 1, buf ^ 1);   // overlap next tile's copy with this tile's math
      WAIT_ASYNC(4);            // this tile's 4 copies done; next tile's may fly
    } else {
      WAIT_ASYNC(0);
    }
    __syncthreads();
    tile_mma(xs[buf], wsm[buf], waveM, waveN, l15, sel, acc);
    __syncthreads();            // before buf is re-filled two steps ahead
  }

  // ---- store: C/D f32 layout — VGPR j: lanes0-15 -> M=j, lanes16-31 -> M=8+j ----
  #pragma unroll
  for (int tm = 0; tm < 2; ++tm) {
    int rowb = n0 + waveM*32 + tm*16 + sel*8;
    #pragma unroll
    for (int tn = 0; tn < 4; ++tn) {
      int col = o0 + waveN*64 + tn*16 + l15;
      #pragma unroll
      for (int j = 0; j < 8; ++j)
        out[(size_t)(rowb + j)*DOUT + col] = acc[tm][tn][j];
    }
  }
}

// ======================================================================
// FALLBACK: convert-in-kernel staging (used when ws can't hold bf16 copies)
// ======================================================================
__global__ __launch_bounds__(256) void main_gemm_kernel(
    const float* __restrict__ x, const float* __restrict__ Wt,
    const unsigned short* __restrict__ Tg, const unsigned short* __restrict__ BcT,
    float* __restrict__ out) {
  __shared__ unsigned short xs[BM * XSTR];
  __shared__ unsigned short wsm[BN * XSTR];

  int tid  = threadIdx.x;
  int lane = tid & 31, wave = tid >> 5;
  int waveM = wave & 3, waveN = wave >> 2;
  int l15 = lane & 15, sel = lane >> 4;
  int n0 = blockIdx.y * BM, o0 = blockIdx.x * BN;

  v8f acc[2][4] = {};

  for (int kt = 0; kt < NKT_DENSE; ++kt) {
    int k0 = kt * BK;
    if (kt + 1 < NKT_DENSE) {
      __builtin_prefetch(&x [(size_t)(n0 + (tid & 127))*DIN + k0 + BK], 0, 1);
      __builtin_prefetch(&Wt[(size_t)(o0 + (tid & 127))*DIN + k0 + BK], 0, 1);
    }
    #pragma unroll
    for (int i = 0; i < 16; ++i) {
      int idx = i*256 + tid;
      int m = idx >> 5, kk = idx & 31;
      xs [m*XSTR + kk] = f2bf(x [(size_t)(n0 + m)*DIN + k0 + kk]);
      wsm[m*XSTR + kk] = f2bf(Wt[(size_t)(o0 + m)*DIN + k0 + kk]);
    }
    __syncthreads();
    tile_mma(xs, wsm, waveM, waveN, l15, sel, acc);
    __syncthreads();
  }

  for (int kt = 0; kt < ER/BK; ++kt) {
    int k0 = kt * BK;
    #pragma unroll
    for (int i = 0; i < 16; ++i) {
      int idx = i*256 + tid;
      int m = idx >> 5, kk = idx & 31;
      xs [m*XSTR + kk] = Tg [(size_t)(n0 + m)*ER + k0 + kk];
      wsm[m*XSTR + kk] = BcT[(size_t)(o0 + m)*ER + k0 + kk];
    }
    __syncthreads();
    tile_mma(xs, wsm, waveM, waveN, l15, sel, acc);
    __syncthreads();
  }

  #pragma unroll
  for (int tm = 0; tm < 2; ++tm) {
    int rowb = n0 + waveM*32 + tm*16 + sel*8;
    #pragma unroll
    for (int tn = 0; tn < 4; ++tn) {
      int col = o0 + waveN*64 + tn*16 + l15;
      #pragma unroll
      for (int j = 0; j < 8; ++j)
        out[(size_t)(rowb + j)*DOUT + col] = acc[tm][tn][j];
    }
  }
}

// ---------------- host launch ----------------
extern "C" void kernel_launch(void* const* d_in, const int* in_sizes, int n_in,
                              void* d_out, int out_size, void* d_ws, size_t ws_size,
                              hipStream_t stream) {
  const float* x   = (const float*)d_in[0];
  const float* wt  = (const float*)d_in[1];
  const float* lA  = (const float*)d_in[2];
  const float* lB  = (const float*)d_in[3];
  const float* gw1 = (const float*)d_in[4];
  const float* gb1 = (const float*)d_in[5];
  const float* lng = (const float*)d_in[6];
  const float* lnb = (const float*)d_in[7];
  const float* gw2 = (const float*)d_in[8];
  const float* gb2 = (const float*)d_in[9];
  float* out = (float*)d_out;

  char* ws = (char*)d_ws;
  float* gates = (float*)(ws);                          // 256 KB
  float* topg  = (float*)(ws + (256u<<10));             //  64 KB
  int*   topi  = (int*)  (ws + (320u<<10));             //  64 KB
  float* wcomb = (float*)(ws + (384u<<10));             // 256 KB
  int*   counts= (int*)  (ws + (640u<<10));             //  tiny
  unsigned short* Tg  = (unsigned short*)(ws + (1u<<20));   // 2 MB  [N_TOK, ER]
  unsigned short* BcT = (unsigned short*)(ws + (3u<<20));   // 1 MB  [DOUT, ER]
  unsigned short* xb  = (unsigned short*)(ws + (4u<<20));   // 64 MB [N_TOK, DIN]
  unsigned short* wb  = (unsigned short*)(ws + (68u<<20));  // 32 MB [DOUT, DIN]
  const size_t WS_FAST = 100ull << 20;

  zero_counts_kernel<<<1, 32, 0, stream>>>(counts);
  gate_kernel <<<N_TOK/8,   256, 0, stream>>>(x, gw1, gb1, lng, lnb, gw2, gb2, gates);
  route_kernel<<<N_TOK/256, 256, 0, stream>>>(gates, topi, topg, counts);
  wcomb_kernel<<<N_TOK/256, 256, 0, stream>>>(topi, topg, counts, wcomb);
  lora_t_kernel<<<N_TOK, 128, 0, stream>>>(x, lA, wcomb, Tg);
  bconvT_kernel<<<(ER*DOUT)/256, 256, 0, stream>>>(lB, BcT);

  dim3 grid(DOUT/BN, N_TOK/BM);   // 32 x 64 blocks
  if (ws_size >= WS_FAST) {
    // one-shot fp32->bf16 (bf16 working set of 96 MB then lives in the 192 MB L2)
    cvt_bf16_kernel<<<(N_TOK*(size_t)DIN)/(256*8), 256, 0, stream>>>(x,  xb);
    cvt_bf16_kernel<<<(DOUT *(size_t)DIN)/(256*8), 256, 0, stream>>>(wt, wb);
    main_gemm_async_kernel<<<grid, 256, 0, stream>>>(xb, wb, Tg, BcT, out);
  } else {
    main_gemm_kernel<<<grid, 256, 0, stream>>>(x, wt, Tg, BcT, out);
  }
}
